// Net_10720238370879
// MI455X (gfx1250) — matrix operand order repaired and checked by reading, not compiled, not run on hardware
//
#include <hip/hip_runtime.h>
#include <hip/hip_bf16.h>
#include <math.h>

// Problem constants (match reference)
#define BB   8
#define TT   2048
#define CC   3
#define EE   256
#define KW   15
#define ND   5
#define BNN  32
#define REPS 1e-5f

typedef _Float16 half_t;
typedef __attribute__((ext_vector_type(16))) _Float16 v16h;
typedef __attribute__((ext_vector_type(8)))  _Float16 v8h;
typedef __attribute__((ext_vector_type(8)))  float    v8f;

// ---------------------------------------------------------------------------
// WMMA fragment loaders (CDNA5 wave32 layouts, CDNA5 ISA §7.12.2)
// A (16x32 f16): lane L -> row r0+(L&15); elems 0..7  = K k0+hi*8+0..7,
//                                         elems 8..15 = K k0+16+hi*8+0..7
// B (32x16 f16): lane L -> col n0+(L&15); same K order as A
// Both become two 16-byte global loads per lane.
// ---------------------------------------------------------------------------
__device__ __forceinline__ v16h load_a16(const half_t* __restrict__ base,
                                         int r0, int k0, int ld) {
  int lane = threadIdx.x & 31;
  const half_t* pp = base + (size_t)(r0 + (lane & 15)) * ld + k0 + ((lane >> 4) << 3);
  v8h lo = *(const v8h*)pp;
  v8h hh = *(const v8h*)(pp + 16);
  v16h r;
#pragma unroll
  for (int i = 0; i < 8; ++i) { r[i] = lo[i]; r[8 + i] = hh[i]; }
  return r;
}

__device__ __forceinline__ v16h load_b16(const half_t* __restrict__ base,
                                         int n0, int k0, int ld) {
  int lane = threadIdx.x & 31;
  const half_t* pp = base + (size_t)(n0 + (lane & 15)) * ld + k0 + ((lane >> 4) << 3);
  v8h lo = *(const v8h*)pp;
  v8h hh = *(const v8h*)(pp + 16);
  v16h r;
#pragma unroll
  for (int i = 0; i < 8; ++i) { r[i] = lo[i]; r[8 + i] = hh[i]; }
  return r;
}

// ---------------------------------------------------------------------------
// RevIN: per (b,c) stats over T from key_in; normalize key & query -> [B,C,T]
// ---------------------------------------------------------------------------
__global__ __launch_bounds__(256) void revin_kernel(
    const float* __restrict__ q_in, const float* __restrict__ k_in,
    const float* __restrict__ rw, const float* __restrict__ rb,
    float* __restrict__ meanv, float* __restrict__ stdv,
    float* __restrict__ qn, float* __restrict__ kn) {
  int bc = blockIdx.x;
  int b = bc / CC, c = bc % CC;
  int tid = threadIdx.x;
  float s1 = 0.f, s2 = 0.f;
  for (int t = tid; t < TT; t += 256) {
    float x = k_in[((size_t)b * TT + t) * CC + c];
    s1 += x; s2 += x * x;
  }
  __shared__ float r1[256], r2[256];
  __shared__ float sm, ss;
  r1[tid] = s1; r2[tid] = s2;
  __syncthreads();
  for (int off = 128; off > 0; off >>= 1) {
    if (tid < off) { r1[tid] += r1[tid + off]; r2[tid] += r2[tid + off]; }
    __syncthreads();
  }
  if (tid == 0) {
    float mean = r1[0] * (1.f / TT);
    float var  = r2[0] * (1.f / TT) - mean * mean;
    float sd   = sqrtf(var + REPS);
    meanv[bc] = mean; stdv[bc] = sd; sm = mean; ss = sd;
  }
  __syncthreads();
  float mean = sm, inv = 1.f / ss, w = rw[c], bv = rb[c];
  for (int t = tid; t < TT; t += 256) {
    float kv = k_in[((size_t)b * TT + t) * CC + c];
    float qv = q_in[((size_t)b * TT + t) * CC + c];
    kn[((size_t)b * CC + c) * TT + t] = (kv - mean) * inv * w + bv;
    qn[((size_t)b * CC + c) * TT + t] = (qv - mean) * inv * w + bv;
  }
}

// ---------------------------------------------------------------------------
// Input conv: [B,C,T] -> relu(conv K=15 dil=1) -> [B,BN,T]
// ---------------------------------------------------------------------------
__global__ __launch_bounds__(256) void conv_in_kernel(
    const float* __restrict__ x, const float* __restrict__ w,
    const float* __restrict__ bias, float* __restrict__ y) {
  int blk = blockIdx.x;
  int b = blk / BNN, oc = blk % BNN;
  __shared__ float ws[CC * KW];
  if (threadIdx.x < CC * KW) ws[threadIdx.x] = w[oc * CC * KW + threadIdx.x];
  __syncthreads();
  float bv = bias[oc];
  const float* xb = x + (size_t)b * CC * TT;
  float* yb = y + ((size_t)b * BNN + oc) * TT;
  for (int t = threadIdx.x; t < TT; t += 256) {
    float acc = bv;
#pragma unroll
    for (int ci = 0; ci < CC; ++ci)
#pragma unroll
      for (int kk = 0; kk < KW; ++kk) {
        int tt = t + kk - 7;
        if ((unsigned)tt < TT) acc += xb[ci * TT + tt] * ws[ci * KW + kk];
      }
    yb[t] = fmaxf(acc, 0.f);
  }
}

// ---------------------------------------------------------------------------
// Dilated mid conv: full [BN,T] activation tile staged in the 320KB WGP LDS
// (256KB activations + 15KB weights for 8 output channels).
// ---------------------------------------------------------------------------
__global__ __launch_bounds__(256) void conv_mid_kernel(
    const float* __restrict__ x, const float* __restrict__ w,
    const float* __restrict__ bias, float* __restrict__ y, int dil) {
  extern __shared__ float smem[];
  float* xs  = smem;              // BN*T
  float* wsm = smem + BNN * TT;   // 8*BN*KW
  int b = blockIdx.x >> 2, g = blockIdx.x & 3;
  const float* xb = x + (size_t)b * BNN * TT;
  for (int i = threadIdx.x; i < BNN * TT; i += 256) xs[i] = xb[i];
  for (int i = threadIdx.x; i < 8 * BNN * KW; i += 256)
    wsm[i] = w[(size_t)g * 8 * BNN * KW + i];
  __syncthreads();
  for (int ol = 0; ol < 8; ++ol) {
    int oc = g * 8 + ol;
    float bv = bias[oc];
    float* yb = y + ((size_t)b * BNN + oc) * TT;
    const float* wrow = wsm + ol * BNN * KW;
    for (int t = threadIdx.x; t < TT; t += 256) {
      float acc = bv;
      for (int kk = 0; kk < KW; ++kk) {
        int tt = t + (kk - 7) * dil;
        if ((unsigned)tt < TT) {
#pragma unroll
          for (int ic = 0; ic < BNN; ++ic)
            acc += xs[ic * TT + tt] * wrow[ic * KW + kk];
        }
      }
      yb[t] = fmaxf(acc, 0.f);
    }
  }
}

// ---------------------------------------------------------------------------
// 1x1 out conv (BN->E) + pack to f16 in WMMA-friendly layouts:
//   mode 0: Q [B,T,E] scaled by E^-0.5 = 1/16
//   mode 1: K [B,T,E]
//   mode 2: V transposed [B,E,T]  (so V B-fragments are contiguous loads)
// ---------------------------------------------------------------------------
__global__ __launch_bounds__(256) void conv_out_pack_kernel(
    const float* __restrict__ h, const float* __restrict__ w,
    const float* __restrict__ bias, half_t* __restrict__ dst, int mode) {
  __shared__ float wsm[EE * BNN];
  __shared__ float bsm[EE];
  for (int i = threadIdx.x; i < EE * BNN; i += 256) wsm[i] = w[i];
  if (threadIdx.x < EE) bsm[threadIdx.x] = bias[threadIdx.x];
  __syncthreads();
  int b = blockIdx.x >> 3;
  int t = ((blockIdx.x & 7) << 8) + threadIdx.x;
  float hr[BNN];
#pragma unroll
  for (int ic = 0; ic < BNN; ++ic) hr[ic] = h[((size_t)b * BNN + ic) * TT + t];
  for (int e = 0; e < EE; ++e) {
    float acc = bsm[e];
#pragma unroll
    for (int ic = 0; ic < BNN; ++ic) acc += hr[ic] * wsm[e * BNN + ic];
    if (mode == 0)      dst[((size_t)b * TT + t) * EE + e] = (half_t)(acc * 0.0625f);
    else if (mode == 1) dst[((size_t)b * TT + t) * EE + e] = (half_t)acc;
    else                dst[((size_t)b * EE + e) * TT + t] = (half_t)acc;
  }
}

// ---------------------------------------------------------------------------
// Flash attention, wave32 WMMA. Each wave owns 16 queries; block = 8 waves
// = 128 queries; grid = B * (T/128).
// S^T = K·Q^T so the softmaxed probabilities land directly in the A-fragment
// layout of the P·V WMMA with zero cross-lane movement.
// ---------------------------------------------------------------------------
__global__ __launch_bounds__(256) void attn_kernel(
    const half_t* __restrict__ Q, const half_t* __restrict__ Kmat,
    const half_t* __restrict__ Vt, float* __restrict__ O) {
  int lane = threadIdx.x & 31;
  int wave = threadIdx.x >> 5;
  int b  = blockIdx.x >> 4;
  int q0 = ((blockIdx.x & 15) << 7) + (wave << 4);
  const half_t* Qb = Q    + (size_t)b * TT * EE;
  const half_t* Kb = Kmat + (size_t)b * TT * EE;
  const half_t* Vb = Vt   + (size_t)b * EE * TT;   // [E][T]

  // Q^T B-fragments for this wave's 16 queries, all of E: kept in VGPRs.
  v16h qf[8];
#pragma unroll
  for (int c = 0; c < 8; ++c) qf[c] = load_b16(Qb, q0, c * 32, EE);

  float m = -3.0e38f, l = 0.f;
  v8f acc[16];
#pragma unroll
  for (int n = 0; n < 16; ++n)
#pragma unroll
    for (int v = 0; v < 8; ++v) acc[n][v] = 0.f;

  int base = (lane >> 4) << 3;   // acc row half owned by this lane group

  for (int j = 0; j < TT; j += 32) {
    if (j + 32 < TT) {
      __builtin_prefetch(Kb + (size_t)(j + 32) * EE, 0, 1);  // global_prefetch_b8
      __builtin_prefetch(Vb + (j + 32), 0, 1);
    }
    // S^T tile: keys j..j+31 (two 16-row A tiles) x 16 queries, accum over E
    v8f s0, s1;
#pragma unroll
    for (int v = 0; v < 8; ++v) { s0[v] = 0.f; s1[v] = 0.f; }
#pragma unroll
    for (int c = 0; c < 8; ++c) {
      v16h a0 = load_a16(Kb, j,      c * 32, EE);
      v16h a1 = load_a16(Kb, j + 16, c * 32, EE);
      s0 = __builtin_amdgcn_wmma_f32_16x16x32_f16(false, a0, false, qf[c],
                                                  (short)0, s0, false, false);
      s1 = __builtin_amdgcn_wmma_f32_16x16x32_f16(false, a1, false, qf[c],
                                                  (short)0, s1, false, false);
    }
    // Online softmax: per lane = one query column, 16 keys in regs + pair half
    float mx = s0[0];
#pragma unroll
    for (int v = 0; v < 8; ++v) { mx = fmaxf(mx, s0[v]); mx = fmaxf(mx, s1[v]); }
    mx = fmaxf(mx, __shfl_xor(mx, 16, 32));
    float mnew  = fmaxf(m, mx);
    float alpha = __expf(m - mnew);
    float rs = 0.f;
    v16h ap;                      // P in A-fragment layout, directly
#pragma unroll
    for (int v = 0; v < 8; ++v) {
      float p0 = __expf(s0[v] - mnew);
      float p1 = __expf(s1[v] - mnew);
      rs += p0 + p1;
      ap[v]     = (half_t)p0;
      ap[8 + v] = (half_t)p1;
    }
    rs += __shfl_xor(rs, 16, 32);
    l = l * alpha + rs;
    m = mnew;
    // rescale accumulator rows (row query = base+v; alpha lives at lane=query)
#pragma unroll
    for (int v = 0; v < 8; ++v) {
      float av = __shfl(alpha, base + v, 32);
#pragma unroll
      for (int n = 0; n < 16; ++n) acc[n][v] *= av;
    }
    // O += P(16x32) · V(32x16) for each of 16 embed subtiles
#pragma unroll
    for (int n = 0; n < 16; ++n) {
      v16h bv = load_b16(Vb, n * 16, j, TT);
      acc[n] = __builtin_amdgcn_wmma_f32_16x16x32_f16(false, ap, false, bv,
                                                      (short)0, acc[n], false, false);
    }
  }
  // epilogue: divide by l and store f32 [B,T,E]
  float* Ob = O + ((size_t)b * TT + q0) * EE;
#pragma unroll
  for (int v = 0; v < 8; ++v) {
    float lv = __shfl(l, base + v, 32);
    float linv = 1.f / lv;
#pragma unroll
    for (int n = 0; n < 16; ++n)
      Ob[(size_t)(base + v) * EE + n * 16 + (lane & 15)] = acc[n][v] * linv;
  }
}

// ---------------------------------------------------------------------------
// Output projection [B,T,E] @ [C,E]^T + bias, then RevIN denorm -> [B,T,C]
// ---------------------------------------------------------------------------
__global__ __launch_bounds__(256) void out_proj_kernel(
    const float* __restrict__ O, const float* __restrict__ ow,
    const float* __restrict__ ob, const float* __restrict__ rw,
    const float* __restrict__ rb, const float* __restrict__ meanv,
    const float* __restrict__ stdv, float* __restrict__ out) {
  int idx = blockIdx.x * 256 + threadIdx.x;
  if (idx >= BB * TT * CC) return;
  int c = idx % CC;
  int t = (idx / CC) % TT;
  int b = idx / (CC * TT);
  const float* orow = O + ((size_t)b * TT + t) * EE;
  const float* wrow = ow + c * EE;
  float acc = ob[c];
#pragma unroll 8
  for (int e = 0; e < EE; ++e) acc += orow[e] * wrow[e];
  out[idx] = ((acc - rb[c]) / rw[c]) * stdv[b * CC + c] + meanv[b * CC + c];
}

// ---------------------------------------------------------------------------
extern "C" void kernel_launch(void* const* d_in, const int* in_sizes, int n_in,
                              void* d_out, int out_size, void* d_ws, size_t ws_size,
                              hipStream_t stream) {
  (void)in_sizes; (void)n_in; (void)out_size; (void)ws_size;
  const float* q_in = (const float*)d_in[0];
  const float* k_in = (const float*)d_in[1];
  const float* nw_in[3]  = {(const float*)d_in[2],  (const float*)d_in[8],  (const float*)d_in[14]};
  const float* nb_in[3]  = {(const float*)d_in[3],  (const float*)d_in[9],  (const float*)d_in[15]};
  const float* nw_mid[3] = {(const float*)d_in[4],  (const float*)d_in[10], (const float*)d_in[16]};
  const float* nb_mid[3] = {(const float*)d_in[5],  (const float*)d_in[11], (const float*)d_in[17]};
  const float* nw_out[3] = {(const float*)d_in[6],  (const float*)d_in[12], (const float*)d_in[18]};
  const float* nb_out[3] = {(const float*)d_in[7],  (const float*)d_in[13], (const float*)d_in[19]};
  const float* out_w   = (const float*)d_in[20];
  const float* out_b   = (const float*)d_in[21];
  const float* revin_w = (const float*)d_in[22];
  const float* revin_b = (const float*)d_in[23];

  // Bump allocator over workspace (256B aligned regions, ~45 MB total)
  char* p = (char*)d_ws;
  auto alloc = [&](size_t bytes) {
    char* r = p;
    p += (bytes + 255) & ~(size_t)255;
    return r;
  };
  float*  meanv = (float*)alloc(BB * CC * sizeof(float));
  float*  stdv  = (float*)alloc(BB * CC * sizeof(float));
  float*  qn    = (float*)alloc((size_t)BB * CC * TT * sizeof(float));
  float*  kn    = (float*)alloc((size_t)BB * CC * TT * sizeof(float));
  float*  h0    = (float*)alloc((size_t)BB * BNN * TT * sizeof(float));
  float*  h1    = (float*)alloc((size_t)BB * BNN * TT * sizeof(float));
  half_t* Qh    = (half_t*)alloc((size_t)BB * TT * EE * sizeof(half_t));
  half_t* Kh    = (half_t*)alloc((size_t)BB * TT * EE * sizeof(half_t));
  half_t* Vth   = (half_t*)alloc((size_t)BB * EE * TT * sizeof(half_t));
  float*  Obuf  = (float*)alloc((size_t)BB * TT * EE * sizeof(float));

  revin_kernel<<<BB * CC, 256, 0, stream>>>(q_in, k_in, revin_w, revin_b,
                                            meanv, stdv, qn, kn);

  const size_t midLds = (size_t)(BNN * TT + 8 * BNN * KW) * sizeof(float); // 271KB < 320KB
  half_t* dsts[3] = {Qh, Kh, Vth};
  for (int net = 0; net < 3; ++net) {
    const float* src = (net == 0) ? qn : kn;   // q-net on qn; k,v-nets on kn
    conv_in_kernel<<<BB * BNN, 256, 0, stream>>>(src, nw_in[net], nb_in[net], h0);
    float* cur = h0; float* nxt = h1;
    for (int i = 0; i < ND; ++i) {
      conv_mid_kernel<<<BB * 4, 256, midLds, stream>>>(
          cur, nw_mid[net] + (size_t)i * BNN * BNN * KW, nb_mid[net] + i * BNN,
          nxt, 2 << i);                        // dilation 2,4,8,16,32
      float* tmp = cur; cur = nxt; nxt = tmp;
    }
    conv_out_pack_kernel<<<BB * 8, 256, 0, stream>>>(cur, nw_out[net],
                                                     nb_out[net], dsts[net], net);
  }

  attn_kernel<<<BB * 16, 256, 0, stream>>>(Qh, Kh, Vth, Obuf);

  out_proj_kernel<<<(BB * TT * CC) / 256, 256, 0, stream>>>(
      Obuf, out_w, out_b, revin_w, revin_b, meanv, stdv, (float*)d_out);
}
